// SparseAttention_352187318824
// MI455X (gfx1250) — matrix operand order repaired and checked by reading, not compiled, not run on hardware
//
#include <hip/hip_runtime.h>

#define DIMX 1024
#define HEADS 16
#define DH 64
#define BLK 64
#define NSEL 8
#define NSEQ 2048
#define NW 32             // key blocks = NSEQ/BLK
#define CD 4096
#define NCTX (NSEL*BLK)   // 512 selected keys per query

typedef __bf16 bf16;
typedef bf16  bf16x16 __attribute__((ext_vector_type(16)));
typedef float floatx8 __attribute__((ext_vector_type(8)));

// ---------------------------------------------------------------------------
// bf16 WMMA GEMM: C[MxN] = act( A[MxK] @ B[KxN] + bias )
// 256 threads (8 waves), block tile 128x128, K-step 32, double-buffered LDS.
// Each wave owns a 32x64 strip: 2 A-fragments x 4 B-fragments = 8 WMMAs per
// K-step, all fragments preloaded so a single s_wait_dscnt covers the burst.
// A staged row-major; B staged transposed so both fragment loads are two
// contiguous ds_load_b128 per lane (CDNA5 16-bit A 16x32 / B 32x16 layouts).
// ---------------------------------------------------------------------------
#define BM 128
#define BN 128
#define BK 32
#define LDA 40
#define LDB 40

__global__ __launch_bounds__(256)
void gemm_bf16_wmma(const bf16* __restrict__ A, const bf16* __restrict__ B,
                    const float* __restrict__ bias,
                    float* __restrict__ Cf, bf16* __restrict__ Cb,
                    int M, int N, int K, int relu)
{
    __shared__ bf16 sA[2][BM][LDA];     // 2 x 128 x 32 (padded)
    __shared__ bf16 sBT[2][BN][LDB];    // 2 x 128 x 32 (transposed, padded)

    const int tid  = threadIdx.x;
    const int lane = tid & 31;
    const int wave = tid >> 5;
    const int wm   = wave & 3;          // M strip 0..3 (32 rows each)
    const int wn   = wave >> 2;         // N half  0..1 (64 cols each)
    const int m0   = blockIdx.y * BM;
    const int n0   = blockIdx.x * BN;

    floatx8 acc[2][4] = {};

    // global staging indices
    const int arow = tid >> 1;           // 0..127
    const int acol = (tid & 1) * 16;     // 0,16
    const int brow = tid >> 3;           // 0..31
    const int bcol = (tid & 7) * 16;     // 0..112
    const bool bok = (n0 + bcol) < N;

    const int nk = K / BK;

    uint4 a0 = {}, a1 = {}, b0 = {}, b1 = {};

    // ---- prologue: stage tile 0 ----
    {
        const bf16* ap = A + (size_t)(m0 + arow) * K + acol;
        a0 = *(const uint4*)(ap);
        a1 = *(const uint4*)(ap + 8);
        if (bok) {
            const bf16* bp = B + (size_t)brow * N + n0 + bcol;
            b0 = *(const uint4*)(bp);
            b1 = *(const uint4*)(bp + 8);
        }
        *(uint4*)&sA[0][arow][acol]     = a0;
        *(uint4*)&sA[0][arow][acol + 8] = a1;
        const bf16* e0 = (const bf16*)&b0;
        const bf16* e1 = (const bf16*)&b1;
        #pragma unroll
        for (int i = 0; i < 8; ++i) sBT[0][bcol + i][brow]     = e0[i];
        #pragma unroll
        for (int i = 0; i < 8; ++i) sBT[0][bcol + 8 + i][brow] = e1[i];
    }

    for (int kt = 0; kt < nk; ++kt) {
        __syncthreads();                 // buf[kt&1] ready
        const int cur = kt & 1;
        const int nxt = cur ^ 1;

        // ---- issue global loads for tile kt+1 (overlap with WMMA burst) ----
        const bool more = (kt + 1) < nk;
        if (more) {
            const int k1 = (kt + 1) * BK;
            const bf16* ap = A + (size_t)(m0 + arow) * K + k1 + acol;
            a0 = *(const uint4*)(ap);
            a1 = *(const uint4*)(ap + 8);
            if (bok) {
                const bf16* bp = B + (size_t)(k1 + brow) * N + n0 + bcol;
                b0 = *(const uint4*)(bp);
                b1 = *(const uint4*)(bp + 8);
            }
            if (kt + 2 < nk)             // pull tile kt+2 toward near caches
                __builtin_prefetch(A + (size_t)(m0 + arow) * K + (kt + 2) * BK + acol, 0, 3);
        }

        // ---- preload ALL fragments, then 8 WMMAs back-to-back ----
        union { bf16x16 v; uint4 u[2]; } af[2], bf[4];
        const int akb = (lane < 16) ? 0 : 8;    // A: lanes 0-15 hold K 0-7/16-23
        const int bkb = (lane < 16) ? 0 : 8;    // B: lanes 0-15 hold K 0-7/16-23
        #pragma unroll
        for (int mi = 0; mi < 2; ++mi) {
            const int am = wm * 32 + mi * 16 + (lane & 15);
            af[mi].u[0] = *(const uint4*)&sA[cur][am][akb];
            af[mi].u[1] = *(const uint4*)&sA[cur][am][akb + 16];
        }
        #pragma unroll
        for (int s = 0; s < 4; ++s) {
            const int bn = wn * 64 + s * 16 + (lane & 15);
            bf[s].u[0] = *(const uint4*)&sBT[cur][bn][bkb];
            bf[s].u[1] = *(const uint4*)&sBT[cur][bn][bkb + 16];
        }
        #pragma unroll
        for (int mi = 0; mi < 2; ++mi)
            #pragma unroll
            for (int s = 0; s < 4; ++s)
                acc[mi][s] = __builtin_amdgcn_wmma_f32_16x16x32_bf16(
                                 false, af[mi].v, false, bf[s].v,
                                 (short)0, acc[mi][s], false, false);

        // ---- stage tile kt+1 into the other buffer ----
        if (more) {
            *(uint4*)&sA[nxt][arow][acol]     = a0;
            *(uint4*)&sA[nxt][arow][acol + 8] = a1;
            const bf16* e0 = (const bf16*)&b0;
            const bf16* e1 = (const bf16*)&b1;
            #pragma unroll
            for (int i = 0; i < 8; ++i) sBT[nxt][bcol + i][brow]     = e0[i];
            #pragma unroll
            for (int i = 0; i < 8; ++i) sBT[nxt][bcol + 8 + i][brow] = e1[i];
        }
    }

    // ---- epilogue: bias / relu / f32 and-or bf16 stores ----
    #pragma unroll
    for (int s = 0; s < 4; ++s) {
        const int gcol = n0 + wn * 64 + s * 16 + (lane & 15);
        if (gcol >= N) continue;
        const float bv = bias ? bias[gcol] : 0.0f;
        #pragma unroll
        for (int mi = 0; mi < 2; ++mi) {
            #pragma unroll
            for (int r = 0; r < 8; ++r) {
                const int grow = m0 + wm * 32 + mi * 16 + r + ((lane >> 4) << 3);
                float v = acc[mi][s][r] + bv;
                if (relu) v = fmaxf(v, 0.0f);
                if (Cf) Cf[(size_t)grow * N + gcol] = v;
                if (Cb) Cb[(size_t)grow * N + gcol] = (bf16)v;
            }
        }
    }
}

// ---------------------------------------------------------------------------
__global__ void cast_f32_bf16(const float* __restrict__ in,
                              bf16* __restrict__ out, long n)
{
    long i = blockIdx.x * 256L + threadIdx.x;
    const long stride = (long)gridDim.x * 256L;
    for (; i < n; i += stride) out[i] = (bf16)in[i];
}

// h[(wi*HEADS+head)][t*DH+d] = kv[(wi*BLK+t)*DIMX + head*DH + d] + pe[t][head][d]
__global__ void build_h(const float* __restrict__ kv, const float* __restrict__ pe,
                        bf16* __restrict__ h)
{
    long i = blockIdx.x * 256L + threadIdx.x;
    const long ntot = (long)NW * HEADS * CD;
    const long stride = (long)gridDim.x * 256L;
    for (; i < ntot; i += stride) {
        const int c    = (int)(i & (CD - 1));
        const int row  = (int)(i >> 12);
        const int head = row & (HEADS - 1);
        const int wi   = row >> 4;
        const int t    = c >> 6;
        const int d    = c & 63;
        const float v = kv[(size_t)(wi * BLK + t) * DIMX + head * DH + d]
                      + pe[(size_t)(t * HEADS + head) * DH + d];
        h[i] = (bf16)v;
    }
}

// RoPE on q and k
__global__ void rope_qk(const float* __restrict__ q, const float* __restrict__ k,
                        const float* __restrict__ fc,
                        float* __restrict__ sq, float* __restrict__ sk)
{
    const int i = blockIdx.x * 256 + threadIdx.x;    // (n, h, j)
    if (i >= NSEQ * HEADS * (DH / 2)) return;
    const int j = i & 31;
    const int h = (i >> 5) & 15;
    const int n = i >> 9;
    const float c = fc[(n * 32 + j) * 2 + 0];
    const float s = fc[(n * 32 + j) * 2 + 1];
    const size_t base = (size_t)n * DIMX + h * DH + 2 * j;
    const float qr = q[base], qi = q[base + 1];
    sq[base]     = qr * c - qi * s;
    sq[base + 1] = qr * s + qi * c;
    const float kr = k[base], ki = k[base + 1];
    sk[base]     = kr * c - ki * s;
    sk[base + 1] = kr * s + ki * c;
}

// One wave per (n, h): 32 lanes == 32 compressed blocks.
// scores -> softmax -> top-8 (iterated shfl argmax) -> compressed = p @ cv
__global__ __launch_bounds__(256)
void cscore_topk_compress(const float* __restrict__ q,
                          const float* __restrict__ ck,
                          const float* __restrict__ cv,
                          float* __restrict__ comp,
                          int* __restrict__ idxout)
{
    const int lane = threadIdx.x & 31;
    const int wave = threadIdx.x >> 5;
    const int pair = blockIdx.x * 8 + wave;
    if (pair >= NSEQ * HEADS) return;
    const int h = pair & 15;
    const int n = pair >> 4;

    float s = 0.0f;
    {
        const float* qp  = q  + (size_t)n * DIMX + h * DH;
        const float* ckp = ck + (size_t)(lane * HEADS + h) * DH;
        #pragma unroll 8
        for (int d = 0; d < DH; ++d) s = fmaf(qp[d], ckp[d], s);
    }
    s *= 0.125f;

    float m = s;
    for (int off = 16; off; off >>= 1) m = fmaxf(m, __shfl_xor(m, off, 32));
    const float e = __expf(s - m);
    float sum = e;
    for (int off = 16; off; off >>= 1) sum += __shfl_xor(sum, off, 32);
    const float p = e / sum;

    float mv = p;
    int myidx = 0;
    for (int it = 0; it < NSEL; ++it) {
        float v = mv; int vi = lane;
        for (int off = 16; off; off >>= 1) {
            const float ov = __shfl_xor(v, off, 32);
            const int   oi = __shfl_xor(vi, off, 32);
            if (ov > v || (ov == v && oi < vi)) { v = ov; vi = oi; }
        }
        if (lane == it) myidx = vi;
        if (lane == vi) mv = -1.0f;           // p >= 0, -1 masks selected
    }
    if (lane < NSEL) idxout[(size_t)pair * NSEL + lane] = myidx;

    float a0 = 0.0f, a1 = 0.0f;
    const int d0 = lane * 2;
    for (int kk = 0; kk < NW; ++kk) {
        const float pk = __shfl(p, kk, 32);
        const float* cvp = cv + (size_t)(kk * HEADS + h) * DH + d0;
        a0 = fmaf(pk, cvp[0], a0);
        a1 = fmaf(pk, cvp[1], a1);
    }
    float* op = comp + (size_t)n * DIMX + h * DH + d0;
    op[0] = a0; op[1] = a1;
}

// One wave per (n, h): attend over the 512 keys of the 8 selected blocks.
#define SAW 4
__global__ __launch_bounds__(SAW * 32)
void selected_attention(const float* __restrict__ sq, const float* __restrict__ sk,
                        const float* __restrict__ v, const int* __restrict__ idx,
                        float* __restrict__ out)
{
    __shared__ float pbuf[SAW][NCTX];
    __shared__ int   kbuf[SAW][NCTX];
    __shared__ float qrow[SAW][DH];

    const int lane = threadIdx.x & 31;
    const int wave = threadIdx.x >> 5;
    const int pair = blockIdx.x * SAW + wave;   // grid sized exactly: never OOB
    if (pair >= NSEQ * HEADS) return;
    const int h = pair & 15;
    const int n = pair >> 4;

    {   // stage query row (per-wave; LDS ops of a wave are in order)
        const float* qp = sq + (size_t)n * DIMX + h * DH;
        qrow[wave][lane]      = qp[lane];
        qrow[wave][lane + 32] = qp[lane + 32];
    }

    const int* myidx = idx + (size_t)pair * NSEL;

    float sc[16];
    float mx = -3.0e38f;
    #pragma unroll
    for (int i = 0; i < 16; ++i) {
        const int j   = lane + 32 * i;
        const int blk = myidx[j >> 6];
        const int key = blk * BLK + (j & 63);
        kbuf[wave][j] = key;
        const float* kp = sk + (size_t)key * DIMX + h * DH;
        float s = 0.0f;
        #pragma unroll 8
        for (int d = 0; d < DH; ++d) s = fmaf(qrow[wave][d], kp[d], s);
        s *= 0.125f;
        sc[i] = s;
        mx = fmaxf(mx, s);
    }
    for (int off = 16; off; off >>= 1) mx = fmaxf(mx, __shfl_xor(mx, off, 32));
    float sum = 0.0f;
    #pragma unroll
    for (int i = 0; i < 16; ++i) { sc[i] = __expf(sc[i] - mx); sum += sc[i]; }
    for (int off = 16; off; off >>= 1) sum += __shfl_xor(sum, off, 32);
    const float inv = 1.0f / sum;
    #pragma unroll
    for (int i = 0; i < 16; ++i) pbuf[wave][lane + 32 * i] = sc[i] * inv;

    // P @ V : two output dims per lane
    const int d0 = lane * 2;
    float a0 = 0.0f, a1 = 0.0f;
    for (int j = 0; j < NCTX; ++j) {
        const float p = pbuf[wave][j];
        const float* vp = v + (size_t)kbuf[wave][j] * DIMX + h * DH + d0;
        a0 = fmaf(p, vp[0], a0);
        a1 = fmaf(p, vp[1], a1);
    }
    float* op = out + (size_t)n * DIMX + h * DH + d0;
    op[0] = a0; op[1] = a1;
}

// y = sigmoid(gate0)*compressed + sigmoid(gate1)*selected, as bf16 for Wo GEMM
__global__ void gate_combine(const float* __restrict__ xc,
                             const float* __restrict__ comp,
                             const float* __restrict__ sel,
                             bf16* __restrict__ y)
{
    const long i = blockIdx.x * 256L + threadIdx.x;
    if (i >= (long)NSEQ * DIMX) return;
    const int n = (int)(i >> 10);
    const int h = (int)(i >> 6) & 15;
    float g0 = xc[n * 32 + h];
    float g1 = xc[n * 32 + 16 + h];
    g0 = 1.0f / (1.0f + __expf(-g0));
    g1 = 1.0f / (1.0f + __expf(-g1));
    y[i] = (bf16)(g0 * comp[i] + g1 * sel[i]);
}

// ---------------------------------------------------------------------------
extern "C" void kernel_launch(void* const* d_in, const int* in_sizes, int n_in,
                              void* d_out, int out_size, void* d_ws, size_t ws_size,
                              hipStream_t stream)
{
    (void)in_sizes; (void)n_in; (void)out_size; (void)ws_size;
    const float* x   = (const float*)d_in[0];
    const float* fc  = (const float*)d_in[1];
    const float* Wq  = (const float*)d_in[2];
    const float* Wk  = (const float*)d_in[3];
    const float* Wv  = (const float*)d_in[4];
    const float* Wo  = (const float*)d_in[5];
    const float* pe  = (const float*)d_in[6];
    const float* kW1 = (const float*)d_in[7];
    const float* kb1 = (const float*)d_in[8];
    const float* kW2 = (const float*)d_in[9];
    const float* kb2 = (const float*)d_in[10];
    const float* vW1 = (const float*)d_in[11];
    const float* vb1 = (const float*)d_in[12];
    const float* vW2 = (const float*)d_in[13];
    const float* vb2 = (const float*)d_in[14];
    const float* Wc  = (const float*)d_in[15];
    const float* bc  = (const float*)d_in[16];

    char* ws = (char*)d_ws;
    size_t off = 0;
    auto alloc = [&](size_t bytes) -> void* {
        void* p = ws + off;
        off = (off + bytes + 255) & ~(size_t)255;
        return p;
    };

    bf16* x_bf   = (bf16*)alloc((size_t)NSEQ * DIMX * 2);
    bf16* Wq_bf  = (bf16*)alloc((size_t)DIMX * DIMX * 2);
    bf16* Wk_bf  = (bf16*)alloc((size_t)DIMX * DIMX * 2);
    bf16* Wv_bf  = (bf16*)alloc((size_t)DIMX * DIMX * 2);
    bf16* Wo_bf  = (bf16*)alloc((size_t)DIMX * DIMX * 2);
    bf16* kW1_bf = (bf16*)alloc((size_t)CD * CD * 2);
    bf16* vW1_bf = (bf16*)alloc((size_t)CD * CD * 2);
    bf16* kW2_bf = (bf16*)alloc((size_t)CD * DH * 2);
    bf16* vW2_bf = (bf16*)alloc((size_t)CD * DH * 2);
    bf16* Wc_bf  = (bf16*)alloc((size_t)DIMX * 2 * HEADS * 2);
    float* qf    = (float*)alloc((size_t)NSEQ * DIMX * 4);
    float* kf    = (float*)alloc((size_t)NSEQ * DIMX * 4);
    float* vf    = (float*)alloc((size_t)NSEQ * DIMX * 4);
    float* xc    = (float*)alloc((size_t)NSEQ * 2 * HEADS * 4);
    bf16* hK     = (bf16*)alloc((size_t)NW * HEADS * CD * 2);
    bf16* hV     = (bf16*)alloc((size_t)NW * HEADS * CD * 2);
    bf16* hk1    = (bf16*)alloc((size_t)NW * HEADS * CD * 2);
    bf16* hv1    = (bf16*)alloc((size_t)NW * HEADS * CD * 2);
    float* ckf   = (float*)alloc((size_t)NW * HEADS * DH * 4);
    float* cvf   = (float*)alloc((size_t)NW * HEADS * DH * 4);
    float* sqf   = (float*)alloc((size_t)NSEQ * DIMX * 4);
    float* skf   = (float*)alloc((size_t)NSEQ * DIMX * 4);
    float* comp  = (float*)alloc((size_t)NSEQ * DIMX * 4);
    float* self  = (float*)alloc((size_t)NSEQ * DIMX * 4);
    int*   idx   = (int*)alloc((size_t)HEADS * NSEQ * NSEL * 4);
    bf16*  y_bf  = (bf16*)alloc((size_t)NSEQ * DIMX * 2);

    auto cast = [&](const float* in, bf16* out, long n) {
        int blocks = (int)((n + 255) / 256);
        if (blocks > 4096) blocks = 4096;
        cast_f32_bf16<<<blocks, 256, 0, stream>>>(in, out, n);
    };
    auto gemm = [&](const bf16* A, const bf16* B, const float* bias,
                    float* Cf, bf16* Cb, int M, int N, int K, int relu) {
        dim3 grid((N + BN - 1) / BN, (M + BM - 1) / BM);
        gemm_bf16_wmma<<<grid, 256, 0, stream>>>(A, B, bias, Cf, Cb, M, N, K, relu);
    };

    cast(x,   x_bf,   (long)NSEQ * DIMX);
    cast(Wq,  Wq_bf,  (long)DIMX * DIMX);
    cast(Wk,  Wk_bf,  (long)DIMX * DIMX);
    cast(Wv,  Wv_bf,  (long)DIMX * DIMX);
    cast(Wo,  Wo_bf,  (long)DIMX * DIMX);
    cast(kW1, kW1_bf, (long)CD * CD);
    cast(vW1, vW1_bf, (long)CD * CD);
    cast(kW2, kW2_bf, (long)CD * DH);
    cast(vW2, vW2_bf, (long)CD * DH);
    cast(Wc,  Wc_bf,  (long)DIMX * 2 * HEADS);

    // q, k, v projections + gate logits
    gemm(x_bf, Wq_bf, nullptr, qf, nullptr, NSEQ, DIMX, DIMX, 0);
    gemm(x_bf, Wk_bf, nullptr, kf, nullptr, NSEQ, DIMX, DIMX, 0);
    gemm(x_bf, Wv_bf, nullptr, vf, nullptr, NSEQ, DIMX, DIMX, 0);
    gemm(x_bf, Wc_bf, bc, xc, nullptr, NSEQ, 2 * HEADS, DIMX, 0);

    // compression MLPs
    build_h<<<2048, 256, 0, stream>>>(kf, pe, hK);
    build_h<<<2048, 256, 0, stream>>>(vf, pe, hV);
    gemm(hK,  kW1_bf, kb1, nullptr, hk1, NW * HEADS, CD, CD, 1);
    gemm(hk1, kW2_bf, kb2, ckf, nullptr, NW * HEADS, DH, CD, 0);
    gemm(hV,  vW1_bf, vb1, nullptr, hv1, NW * HEADS, CD, CD, 1);
    gemm(hv1, vW2_bf, vb2, cvf, nullptr, NW * HEADS, DH, CD, 0);

    // compressed attention + top-8 selection
    cscore_topk_compress<<<(NSEQ * HEADS) / 8, 256, 0, stream>>>(qf, ckf, cvf, comp, idx);

    // RoPE then selected attention over the 8 chosen blocks
    rope_qk<<<(NSEQ * HEADS * (DH / 2)) / 256, 256, 0, stream>>>(qf, kf, fc, sqf, skf);
    selected_attention<<<(NSEQ * HEADS) / SAW, SAW * 32, 0, stream>>>(sqf, skf, vf, idx, self);

    // gate, combine, output projection (writes d_out, f32)
    gate_combine<<<(NSEQ * DIMX) / 256, 256, 0, stream>>>(xc, comp, self, y_bf);
    gemm(y_bf, Wo_bf, nullptr, (float*)d_out, nullptr, NSEQ, DIMX, DIMX, 0);
}
